// MultiDomainHyperViT_73375221284960
// MI455X (gfx1250) — matrix operand order repaired and checked by reading, not compiled, not run on hardware
//
#include <hip/hip_runtime.h>
#include <hip/hip_bf16.h>
#include <math.h>

// ---------------- constants ----------------
#define TOK     1184      // 32 * 37 tokens
#define DIMC    192
#define SEQL    37
#define BATCH   32
#define NPATCH  36
#define QKVD    576
#define EXPD    768
#define NEXP    8
#define MEMH    8
#define MEMDH   24

// ---------------- bf16 helpers / WMMA types ----------------
typedef __bf16 bf16x16 __attribute__((ext_vector_type(16)));
typedef float  f32x8   __attribute__((ext_vector_type(8)));

union Frag  { bf16x16 v; uint4 q[2]; };
union Pack8 { unsigned short s[8]; uint4 q; };

__device__ __forceinline__ unsigned short f2bf(float f) {
    unsigned u = __builtin_bit_cast(unsigned, f);
    unsigned r = u + 0x7FFFu + ((u >> 16) & 1u);   // round-to-nearest-even
    return (unsigned short)(r >> 16);
}

__device__ __forceinline__ float gelu_exact(float x) {
    return 0.5f * x * (1.0f + erff(x * 0.70710678118654752f));
}

// ---------------- WMMA GEMM ----------------
// C[M,N] = act(A[M,K]_f32 * B[K,N]_bf16 + bias) (+ Res)
// 128x64 tile per 256-thread block; 8 waves in 4(m) x 2(n) grid; each wave
// computes a 32x32 sub-tile = 2 A-frags x 2 B-frags = 4 WMMAs per K-step.
__global__ __launch_bounds__(256) void k_gemm(
    const float* __restrict__ A, const unsigned short* __restrict__ Bw,
    const float* __restrict__ bias, const float* __restrict__ Res,
    float* __restrict__ C, int M, int N, int K, int act)
{
    __shared__ __align__(16) unsigned short As[128][40];  // [m][k], 80B pitch
    __shared__ __align__(16) unsigned short Bs[64][40];   // [n][k], 80B pitch

    const int tid  = threadIdx.x;
    const int lane = tid & 31, wid = tid >> 5;
    const int bm = blockIdx.y * 128, bn = blockIdx.x * 64;
    const int wm = (wid & 3) * 32, wn = (wid >> 2) * 32;
    const int hi = lane >> 4, lr = lane & 15;

    f32x8 acc[2][2] = {};

    const int srow = tid >> 2;            // staging row 0..63
    const int scol = (tid & 3) * 8;       // staging col {0,8,16,24}

    for (int k0 = 0; k0 < K; k0 += 32) {
        // ---- stage A tile: 128 rows x 32 k (convert f32 -> bf16) ----
        #pragma unroll
        for (int half = 0; half < 2; half++) {
            int r  = srow + half * 64;
            int gm = bm + r;
            Pack8 pa;
            if (gm < M) {
                const float* src = A + (size_t)gm * K + (k0 + scol);
                #pragma unroll
                for (int j = 0; j < 8; j++) pa.s[j] = f2bf(src[j]);
                if (k0 + 32 < K) __builtin_prefetch((const void*)(src + 32), 0, 1);
            } else {
                #pragma unroll
                for (int j = 0; j < 8; j++) pa.s[j] = 0;
            }
            *(uint4*)&As[r][scol] = pa.q;
        }
        // ---- stage B tile: 64 cols x 32 k (transpose [k][n] -> [n][k]) ----
        {
            Pack8 pb;
            int gn = bn + srow;
            if (gn < N) {
                #pragma unroll
                for (int j = 0; j < 8; j++)
                    pb.s[j] = Bw[(size_t)(k0 + scol + j) * N + gn];
            } else {
                #pragma unroll
                for (int j = 0; j < 8; j++) pb.s[j] = 0;
            }
            *(uint4*)&Bs[srow][scol] = pb.q;
        }
        __syncthreads();

        // ---- build fragments per ISA VGPR layouts (wave32) ----
        Frag af[2], bfr[2];
        #pragma unroll
        for (int mf = 0; mf < 2; mf++) {
            const uint4* ap = (const uint4*)&As[wm + mf * 16 + lr][0];
            af[mf].q[0] = ap[hi];          // K = hi*8 .. +7
            af[mf].q[1] = ap[hi + 2];      // K = 16+hi*8 .. +7
        }
        #pragma unroll
        for (int nf = 0; nf < 2; nf++) {
            const uint4* bp = (const uint4*)&Bs[wn + nf * 16 + lr][0];
            bfr[nf].q[0] = bp[hi];         // K = hi*8 .. +7
            bfr[nf].q[1] = bp[hi + 2];     // K = 16+hi*8 .. +7
        }
        #pragma unroll
        for (int mf = 0; mf < 2; mf++)
            #pragma unroll
            for (int nf = 0; nf < 2; nf++)
                acc[mf][nf] = __builtin_amdgcn_wmma_f32_16x16x32_bf16(
                    false, af[mf].v, false, bfr[nf].v, (short)0, acc[mf][nf], false, false);
        __syncthreads();
    }

    // ---- epilogue: bias -> act -> residual ----
    #pragma unroll
    for (int mf = 0; mf < 2; mf++) {
        #pragma unroll
        for (int nf = 0; nf < 2; nf++) {
            const int nn = bn + wn + nf * 16 + lr;
            if (nn < N) {
                #pragma unroll
                for (int g = 0; g < 8; g++) {
                    int m = bm + wm + mf * 16 + hi * 8 + g;
                    if (m < M) {
                        float v = acc[mf][nf][g];
                        if (bias) v += bias[nn];
                        if (act == 1) v = gelu_exact(v);
                        if (Res) v += Res[(size_t)m * N + nn];
                        C[(size_t)m * N + nn] = v;
                    }
                }
            }
        }
    }
}

// ---------------- weight materialization ----------------
// PHM: W[k][n] (bf16), n = a*oc + i, k = b*ic + j, W = sum_h A[h,a,b]*S[h,i,j]
__global__ void k_phm_weight(const float* __restrict__ A, const float* __restrict__ S,
                             unsigned short* __restrict__ W, int oc, int ic)
{
    int N = 3 * oc, K = 3 * ic;
    int idx = blockIdx.x * 256 + threadIdx.x;
    if (idx >= N * K) return;
    int k = idx / N, n = idx % N;
    int a = n / oc, i = n % oc, b = k / ic, j = k % ic;
    float s = A[0 * 9 + a * 3 + b] * S[(size_t)0 * oc * ic + i * ic + j]
            + A[1 * 9 + a * 3 + b] * S[(size_t)1 * oc * ic + i * ic + j]
            + A[2 * 9 + a * 3 + b] * S[(size_t)2 * oc * ic + i * ic + j];
    W[idx] = f2bf(s);
}

__global__ void k_convert(const float* __restrict__ in, unsigned short* __restrict__ out, int n)
{
    int i = blockIdx.x * 256 + threadIdx.x;
    if (i < n) out[i] = f2bf(in[i]);
}

// in f32 [R][C] (row=out, col=in) -> out bf16 [C][R]
__global__ void k_transpose_convert(const float* __restrict__ in, unsigned short* __restrict__ out,
                                    int R, int C)
{
    int i = blockIdx.x * 256 + threadIdx.x;
    if (i >= R * C) return;
    int r = i / C, c = i % C;
    out[(size_t)c * R + r] = f2bf(in[i]);
}

// patch-embed weight: Wp[k=ci*256+kh*16+kw][c=a*64+i] = sum_h A[h,a,ci]*F[h,i,kh*16+kw]
__global__ void k_patch_weight(const float* __restrict__ A, const float* __restrict__ F,
                               unsigned short* __restrict__ Wp)
{
    int idx = blockIdx.x * 256 + threadIdx.x;
    if (idx >= 768 * 192) return;
    int k = idx / 192, c = idx % 192;
    int ci = k / 256, r = k % 256, a = c / 64, i = c % 64;
    float s = A[0 * 9 + a * 3 + ci] * F[(size_t)0 * 16384 + i * 256 + r]
            + A[1 * 9 + a * 3 + ci] * F[(size_t)1 * 16384 + i * 256 + r]
            + A[2 * 9 + a * 3 + ci] * F[(size_t)2 * 16384 + i * 256 + r];
    Wp[idx] = f2bf(s);
}

// ---------------- patch embed ----------------
__global__ void k_im2col(const float* __restrict__ x, float* __restrict__ out)
{
    int idx = blockIdx.x * 256 + threadIdx.x;
    if (idx >= 1152 * 768) return;
    int p = idx / 768, col = idx % 768;
    int bb = p / 36, pp = p % 36, ph = pp / 6, pw = pp % 6;
    int ci = col / 256, rr = col % 256, kh = rr / 16, kw = rr % 16;
    out[idx] = x[(((size_t)bb * 3 + ci) * 96 + (ph * 16 + kh)) * 96 + (pw * 16 + kw)];
}

__global__ void k_assemble(const float* __restrict__ P, const float* __restrict__ cls,
                           const float* __restrict__ pe, const int* __restrict__ dom,
                           float* __restrict__ X)
{
    int idx = blockIdx.x * 256 + threadIdx.x;
    if (idx >= TOK * DIMC) return;
    int t = idx / DIMC, c = idx % DIMC;
    int bb = t / SEQL, s = t % SEQL;
    float v = pe[s * DIMC + c];
    if (s == 0) v += cls[(size_t)dom[0] * DIMC + c];
    else        v += P[(size_t)(bb * NPATCH + (s - 1)) * DIMC + c];
    X[idx] = v;
}

// ---------------- layernorm (one token per block, 192 threads) ----------------
__global__ void k_layernorm(const float* __restrict__ X, const float* __restrict__ g,
                            const float* __restrict__ bb, float* __restrict__ Y)
{
    int t = blockIdx.x, c = threadIdx.x;
    __shared__ float smem[192];
    __shared__ float red[32];
    __shared__ float stat[2];
    float x = X[(size_t)t * DIMC + c];
    smem[c] = x; __syncthreads();
    if (c < 32) { float s = 0; for (int j = 0; j < 6; j++) s += smem[c + 32 * j]; red[c] = s; }
    __syncthreads();
    if (c == 0) { float s = 0; for (int j = 0; j < 32; j++) s += red[j]; stat[0] = s / 192.0f; }
    __syncthreads();
    float m = stat[0], d = x - m;
    smem[c] = d * d; __syncthreads();
    if (c < 32) { float s = 0; for (int j = 0; j < 6; j++) s += smem[c + 32 * j]; red[c] = s; }
    __syncthreads();
    if (c == 0) { float s = 0; for (int j = 0; j < 32; j++) s += red[j]; stat[1] = s / 192.0f; }
    __syncthreads();
    Y[(size_t)t * DIMC + c] = d * rsqrtf(stat[1] + 1e-5f) * g[c] + bb[c];
}

// ---------------- quantum attention (one batch per block) ----------------
__global__ void k_qattn(const float* __restrict__ QKV, const float* __restrict__ supM,
                        const float* __restrict__ entM, float* __restrict__ Out)
{
    int b = blockIdx.x, tid = threadIdx.x;
    __shared__ float Lg[3][37][37];
    // logits
    for (int idx = tid; idx < 3 * 37 * 37; idx += 256) {
        int h = idx / 1369, r = idx % 1369, n = r / 37, m = r % 37;
        const float* q = QKV + (size_t)(b * 37 + n) * QKVD + h * 64;
        const float* k = QKV + (size_t)(b * 37 + m) * QKVD + 192 + h * 64;
        float s = 0;
        for (int d = 0; d < 64; d++) s += q[d] * k[d];
        Lg[h][n][m] = s * 0.125f;
    }
    __syncthreads();
    // sup / ent mixing across heads + tanh
    for (int idx = tid; idx < 37 * 37; idx += 256) {
        int n = idx / 37, m = idx % 37;
        float a0 = Lg[0][n][m], a1 = Lg[1][n][m], a2 = Lg[2][n][m];
        float sup[3], e[3];
        for (int i = 0; i < 3; i++)
            sup[i] = supM[i * 3 + 0] * a0 + supM[i * 3 + 1] * a1 + supM[i * 3 + 2] * a2;
        for (int i = 0; i < 3; i++) {
            float t2 = entM[i * 3 + 0] * sup[0] + entM[i * 3 + 1] * sup[1] + entM[i * 3 + 2] * sup[2];
            e[i] = tanhf(sup[i] + t2);
        }
        Lg[0][n][m] = e[0]; Lg[1][n][m] = e[1]; Lg[2][n][m] = e[2];
    }
    __syncthreads();
    // softmax per (h, n)
    for (int idx = tid; idx < 3 * 37; idx += 256) {
        int h = idx / 37, n = idx % 37;
        float mx = -1e30f;
        for (int m = 0; m < 37; m++) mx = fmaxf(mx, Lg[h][n][m]);
        float s = 0;
        for (int m = 0; m < 37; m++) { float e = expf(Lg[h][n][m] - mx); Lg[h][n][m] = e; s += e; }
        float inv = 1.0f / s;
        for (int m = 0; m < 37; m++) Lg[h][n][m] *= inv;
    }
    __syncthreads();
    // attn @ v
    for (int idx = tid; idx < 37 * DIMC; idx += 256) {
        int n = idx / DIMC, c = idx % DIMC, h = c >> 6, d = c & 63;
        float s = 0;
        for (int m = 0; m < 37; m++)
            s += Lg[h][n][m] * QKV[(size_t)(b * 37 + m) * QKVD + 384 + h * 64 + d];
        Out[(size_t)(b * 37 + n) * DIMC + c] = s;
    }
}

// ---------------- memory attention (one (batch, head) per block) ----------------
__global__ void k_mem_attn(const float* __restrict__ Q, const float* __restrict__ Kb,
                           const float* __restrict__ Vb, float* __restrict__ Out, int Mr)
{
    int b = blockIdx.x, h = blockIdx.y, tid = threadIdx.x;
    __shared__ float Lg[37 * 256];
    const float scale = 0.2041241452319315f;   // 1/sqrt(24)
    for (int idx = tid; idx < 37 * Mr; idx += 256) {
        int n = idx / Mr, m = idx % Mr;
        const float* q = Q  + (size_t)(b * 37 + n) * DIMC + h * MEMDH;
        const float* k = Kb + (size_t)m * DIMC + h * MEMDH;
        float s = 0;
        for (int d = 0; d < MEMDH; d++) s += q[d] * k[d];
        Lg[n * Mr + m] = s * scale;
    }
    __syncthreads();
    for (int n = tid; n < 37; n += 256) {
        float mx = -1e30f;
        for (int m = 0; m < Mr; m++) mx = fmaxf(mx, Lg[n * Mr + m]);
        float s = 0;
        for (int m = 0; m < Mr; m++) { float e = expf(Lg[n * Mr + m] - mx); Lg[n * Mr + m] = e; s += e; }
        float inv = 1.0f / s;
        for (int m = 0; m < Mr; m++) Lg[n * Mr + m] *= inv;
    }
    __syncthreads();
    for (int idx = tid; idx < 37 * MEMDH; idx += 256) {
        int n = idx / MEMDH, d = idx % MEMDH;
        float s = 0;
        for (int m = 0; m < Mr; m++) s += Lg[n * Mr + m] * Vb[(size_t)m * DIMC + h * MEMDH + d];
        Out[(size_t)(b * 37 + n) * DIMC + h * MEMDH + d] = s;
    }
}

// ---------------- MoE router: top-2 softmax weights ----------------
__global__ void k_router(const float* __restrict__ X, const float* __restrict__ rW,
                         const float* __restrict__ rb, const float* __restrict__ dr,
                         const int* __restrict__ dom, float* __restrict__ CW)
{
    int t = blockIdx.x * 256 + threadIdx.x;
    if (t >= TOK) return;
    int dmi = dom[0];
    float lg[8];
    for (int e = 0; e < 8; e++) {
        float s = rb[e] + dr[dmi * 8 + e];
        for (int c = 0; c < DIMC; c++) s += X[(size_t)t * DIMC + c] * rW[c * 8 + e];
        lg[e] = s;
    }
    int i0 = 0;
    for (int e = 1; e < 8; e++) if (lg[e] > lg[i0]) i0 = e;
    int i1 = -1;
    for (int e = 0; e < 8; e++) { if (e == i0) continue; if (i1 < 0 || lg[e] > lg[i1]) i1 = e; }
    float ex = expf(lg[i1] - lg[i0]);
    float w0 = 1.0f / (1.0f + ex);
    float w1 = ex * w0;
    for (int e = 0; e < 8; e++) CW[t * 8 + e] = 0.0f;
    CW[t * 8 + i0] = w0;
    CW[t * 8 + i1] = w1;
}

__global__ void k_moe_accum(float* __restrict__ out, const float* __restrict__ eo,
                            const float* __restrict__ cw, int e)
{
    int i = blockIdx.x * 256 + threadIdx.x;
    if (i >= TOK * DIMC) return;
    out[i] += cw[(i / DIMC) * 8 + e] * eo[i];
}

// ---------------- elementwise ----------------
__global__ void k_zero(float* __restrict__ y, int n)
{ int i = blockIdx.x * 256 + threadIdx.x; if (i < n) y[i] = 0.0f; }

__global__ void k_add(float* __restrict__ y, const float* __restrict__ x, int n)
{ int i = blockIdx.x * 256 + threadIdx.x; if (i < n) y[i] += x[i]; }

__global__ void k_add3(float* __restrict__ y, const float* __restrict__ a,
                       const float* __restrict__ b, const float* __restrict__ c, int n)
{ int i = blockIdx.x * 256 + threadIdx.x; if (i < n) y[i] += a[i] + b[i] + c[i]; }

__global__ void k_extract_cls(const float* __restrict__ X, float* __restrict__ C)
{
    int i = blockIdx.x * 256 + threadIdx.x;
    if (i >= BATCH * DIMC) return;
    int b = i / DIMC, c = i % DIMC;
    C[i] = X[(size_t)(b * SEQL) * DIMC + c];
}

// tiny-N dense linear (final head)
__global__ void k_small_linear(const float* __restrict__ A, const float* __restrict__ W,
                               const float* __restrict__ b, float* __restrict__ C,
                               int M, int N, int K)
{
    int idx = blockIdx.x * 256 + threadIdx.x;
    if (idx >= M * N) return;
    int m = idx / N, n = idx % N;
    float s = b[n];
    for (int k = 0; k < K; k++) s += A[(size_t)m * K + k] * W[(size_t)k * N + n];
    C[idx] = s;
}

// ---------------- host orchestration ----------------
static inline void gemm(hipStream_t st, const float* A, const unsigned short* B,
                        const float* bias, const float* res, float* C,
                        int M, int N, int K, int act)
{
    dim3 g((N + 63) / 64, (M + 127) / 128);
    k_gemm<<<g, dim3(256), 0, st>>>(A, B, bias, res, C, M, N, K, act);
}

#define L1D(kern, total, ...) kern<<<dim3(((total) + 255) / 256), dim3(256), 0, stream>>>(__VA_ARGS__)

extern "C" void kernel_launch(void* const* d_in, const int* in_sizes, int n_in,
                              void* d_out, int out_size, void* d_ws, size_t ws_size,
                              hipStream_t stream)
{
    (void)in_sizes; (void)n_in; (void)out_size; (void)ws_size;

    auto F = [&](int i) -> const float* { return (const float*)d_in[i]; };

    // -------- input indexing (flattened setup_inputs dict order) --------
    const int IN_X = 0, P_A = 1, P_F = 2, P_B = 3, IN_PE = 4, IN_CLS = 5;
    const int BLK0 = 6, BSTRIDE = 78;
    const int POST = BLK0 + 12 * BSTRIDE;            // 942
    const int UNI1W = POST + 0, UNI1B = POST + 1, UNI2W = POST + 2, UNI2B = POST + 3;
    const int HEAD1W = POST + 4, HEAD1B = POST + 5, HEAD2W = POST + 6, HEAD2B = POST + 7;
    const int IN_DOM = POST + 8;
    const int* dom = (const int*)d_in[IN_DOM];

    // -------- workspace carving --------
    char* wp = (char*)d_ws;
    auto carve = [&](size_t bytes) -> void* {
        void* r = (void*)wp;
        wp += (bytes + 255) & ~(size_t)255;
        return r;
    };
    // bf16 weights (per-block regions reused across blocks)
    unsigned short* wPatch = (unsigned short*)carve(768 * 192 * 2);
    unsigned short* wQKV   = (unsigned short*)carve(192 * 576 * 2);
    unsigned short* wProj  = (unsigned short*)carve(192 * 192 * 2);
    unsigned short* wUp[NEXP], *wDown[NEXP];
    for (int e = 0; e < NEXP; e++) {
        wUp[e]   = (unsigned short*)carve(192 * 768 * 2);
        wDown[e] = (unsigned short*)carve(768 * 192 * 2);
    }
    unsigned short *wMq[3], *wMk[3], *wMv[3], *wMo[3], *wMp[3];
    for (int l = 0; l < 3; l++) {
        wMq[l] = (unsigned short*)carve(192 * 192 * 2);
        wMk[l] = (unsigned short*)carve(192 * 192 * 2);
        wMv[l] = (unsigned short*)carve(192 * 192 * 2);
        wMo[l] = (unsigned short*)carve(192 * 192 * 2);
        wMp[l] = (unsigned short*)carve(192 * 192 * 2);
    }
    unsigned short* wU1 = (unsigned short*)carve(192 * 768 * 2);
    unsigned short* wU2 = (unsigned short*)carve(768 * 192 * 2);
    unsigned short* wH1 = (unsigned short*)carve(192 * 768 * 2);
    // f32 activations
    float* imcol = (float*)carve((size_t)1152 * 768 * 4);
    float* pOut  = (float*)carve((size_t)1152 * 192 * 4);
    float* X     = (float*)carve((size_t)TOK * DIMC * 4);
    float* LN    = (float*)carve((size_t)TOK * DIMC * 4);
    float* QKV   = (float*)carve((size_t)TOK * QKVD * 4);
    float* AOut  = (float*)carve((size_t)TOK * DIMC * 4);
    float* TmpA  = (float*)carve((size_t)TOK * EXPD * 4);
    float* TmpB  = (float*)carve((size_t)TOK * DIMC * 4);
    float* MoE   = (float*)carve((size_t)TOK * DIMC * 4);
    float* CW    = (float*)carve((size_t)TOK * 8 * 4);
    float* Qb    = (float*)carve((size_t)TOK * DIMC * 4);
    float* Kb    = (float*)carve((size_t)256 * DIMC * 4);
    float* Vb    = (float*)carve((size_t)256 * DIMC * 4);
    float* Mb[3];
    for (int l = 0; l < 3; l++) Mb[l] = (float*)carve((size_t)TOK * DIMC * 4);
    float* CLSf  = (float*)carve((size_t)BATCH * DIMC * 4);
    float* U1    = (float*)carve((size_t)BATCH * EXPD * 4);
    float* U2    = (float*)carve((size_t)BATCH * DIMC * 4);
    float* H1    = (float*)carve((size_t)BATCH * EXPD * 4);

    // -------- global weights (once) --------
    L1D(k_patch_weight, 768 * 192, F(P_A), F(P_F), wPatch);
    L1D(k_convert, 192 * 768, F(UNI1W), wU1, 192 * 768);
    L1D(k_convert, 768 * 192, F(UNI2W), wU2, 768 * 192);
    L1D(k_convert, 192 * 768, F(HEAD1W), wH1, 192 * 768);

    // -------- patch embed --------
    L1D(k_im2col, 1152 * 768, F(IN_X), imcol);
    gemm(stream, imcol, wPatch, F(P_B), nullptr, pOut, 1152, 192, 768, 0);
    L1D(k_assemble, TOK * DIMC, pOut, F(IN_CLS), F(IN_PE), dom, X);

    // -------- transformer blocks --------
    for (int blk = 0; blk < 12; blk++) {
        const int o = BLK0 + blk * BSTRIDE;
        const int QKV_A = o + 0, QKV_S = o + 1, QKV_Bv = o + 2;
        const int PRJ_A = o + 3, PRJ_S = o + 4, PRJ_Bv = o + 5;
        const int SUP = o + 6, ENT = o + 7;
        const int RW = o + 8, RB = o + 9, DRT = o + 10;
        const int EXB = o + 11;          // + e*5: upA, upS, upB, dnW, dnB
        const int MMB = o + 51;          // + l*7: bank, inW, inB, outW, outB, prW, prB
        const int LN1G = o + 72, LN1B = o + 73, LN2G = o + 74, LN2B = o + 75;
        const int LN3G = o + 76, LN3B = o + 77;

        // --- materialize block weights (bf16, [K][N]) ---
        L1D(k_phm_weight, 192 * 576, F(QKV_A), F(QKV_S), wQKV, 192, 64);
        L1D(k_phm_weight, 192 * 192, F(PRJ_A), F(PRJ_S), wProj, 64, 64);
        for (int e = 0; e < NEXP; e++) {
            const int eb = EXB + e * 5;
            L1D(k_phm_weight, 192 * 768, F(eb + 0), F(eb + 1), wUp[e], 256, 64);
            L1D(k_convert, 768 * 192, F(eb + 3), wDown[e], 768 * 192);
        }
        for (int l = 0; l < 3; l++) {
            const int mb = MMB + l * 7;
            const float* inW = F(mb + 1);
            L1D(k_transpose_convert, 192 * 192, inW,                 wMq[l], 192, 192);
            L1D(k_transpose_convert, 192 * 192, inW + 192 * 192,     wMk[l], 192, 192);
            L1D(k_transpose_convert, 192 * 192, inW + 2 * 192 * 192, wMv[l], 192, 192);
            L1D(k_transpose_convert, 192 * 192, F(mb + 3), wMo[l], 192, 192);
            L1D(k_convert, 192 * 192, F(mb + 5), wMp[l], 192 * 192);
        }

        // --- attention: x = x + proj(attn(ln1(x))) ---
        k_layernorm<<<dim3(TOK), dim3(192), 0, stream>>>(X, F(LN1G), F(LN1B), LN);
        gemm(stream, LN, wQKV, F(QKV_Bv), nullptr, QKV, TOK, QKVD, DIMC, 0);
        k_qattn<<<dim3(BATCH), dim3(256), 0, stream>>>(QKV, F(SUP), F(ENT), AOut);
        gemm(stream, AOut, wProj, F(PRJ_Bv), X, X, TOK, DIMC, DIMC, 0);

        // --- MoE: x = x + sum_e cw_e * down_e(gelu(up_e(ln2(x)))) ---
        k_layernorm<<<dim3(TOK), dim3(192), 0, stream>>>(X, F(LN2G), F(LN2B), LN);
        L1D(k_router, TOK, LN, F(RW), F(RB), F(DRT), dom, CW);
        L1D(k_zero, TOK * DIMC, MoE, TOK * DIMC);
        for (int e = 0; e < NEXP; e++) {
            const int eb = EXB + e * 5;
            gemm(stream, LN, wUp[e], F(eb + 2), nullptr, TmpA, TOK, EXPD, DIMC, 1);
            gemm(stream, TmpA, wDown[e], F(eb + 4), nullptr, TmpB, TOK, DIMC, EXPD, 0);
            L1D(k_moe_accum, TOK * DIMC, MoE, TmpB, CW, e);
        }
        L1D(k_add, TOK * DIMC, X, MoE, TOK * DIMC);

        // --- fractal memory: x = x + a0 + c1 + d2 ---
        for (int l = 0; l < 3; l++) {
            const int mb = MMB + l * 7;
            const int Mr = 64 << l;
            const float* cur = (l == 0) ? X : Mb[l - 1];
            gemm(stream, F(mb + 0), wMk[l], F(mb + 2) + 192, nullptr, Kb, Mr, DIMC, DIMC, 0);
            gemm(stream, F(mb + 0), wMv[l], F(mb + 2) + 384, nullptr, Vb, Mr, DIMC, DIMC, 0);
            gemm(stream, cur, wMq[l], F(mb + 2), nullptr, Qb, TOK, DIMC, DIMC, 0);
            k_mem_attn<<<dim3(BATCH, MEMH), dim3(256), 0, stream>>>(Qb, Kb, Vb, AOut, Mr);
            gemm(stream, AOut, wMo[l], F(mb + 4), nullptr, TmpB, TOK, DIMC, DIMC, 0);
            gemm(stream, TmpB, wMp[l], F(mb + 6), nullptr, Mb[l], TOK, DIMC, DIMC, 0);
        }
        L1D(k_add3, TOK * DIMC, X, Mb[0], Mb[1], Mb[2], TOK * DIMC);

        // --- ln3 (in place) ---
        k_layernorm<<<dim3(TOK), dim3(192), 0, stream>>>(X, F(LN3G), F(LN3B), X);
    }

    // -------- head --------
    L1D(k_extract_cls, BATCH * DIMC, X, CLSf);
    gemm(stream, CLSf, wU1, F(UNI1B), nullptr, U1, BATCH, EXPD, DIMC, 1);
    gemm(stream, U1, wU2, F(UNI2B), nullptr, U2, BATCH, DIMC, EXPD, 0);
    gemm(stream, U2, wH1, F(HEAD1B), nullptr, H1, BATCH, EXPD, DIMC, 1);
    L1D(k_small_linear, BATCH * 6, H1, F(HEAD2W), F(HEAD2B), (float*)d_out, BATCH, 6, EXPD);
}
